// AttentiveTransformer_41042707481070
// MI455X (gfx1250) — matrix operand order repaired and checked by reading, not compiled, not run on hardware
//
#include <hip/hip_runtime.h>
#include <hip/hip_bf16.h>

// ---------------------------------------------------------------------------
// AttentiveTransformer fused pipeline for MI455X (gfx1250, wave32):
//   kernel 1: x = feat @ W^T  (bf16 WMMA, f32 accum), fused GhostBN (VBS=128
//             == block tile M) + gamma/beta + *priors  -> writes d_out
//            software-pipelined: global loads of tile k+1 fly under the
//            WMMAs of tile k; cheap packed fp32->bf16 staging (v_perm_b32).
//   kernel 2: in-place sparsemax per row via tau-bisection + exact finish
// ---------------------------------------------------------------------------

typedef __attribute__((ext_vector_type(16))) __bf16 v16bf;
typedef __attribute__((ext_vector_type(8)))  float  v8f;

#define NROWS 16384
#define IN_F  2048
#define OUT_F 1024
#define VBS   128
#define TK    64
#define LDSU  36   // uints per LDS row: 72 bf16 = 64 data + 8 pad (start banks 36r mod 64 distinct)

// fp32 -> packed bf16 x2.
// Preferred: single v_cvt_pk_bf16_f32 (if toolchain exposes it).
// Fallback: round-to-nearest (+0x8000) and byte-pack the two high halves with
// one v_perm_b32 -> 3 VALU per pair (vs ~10 for the RNE emulation clang emits).
__device__ __forceinline__ unsigned int pack2_bf16(float a, float b) {
#if __has_builtin(__builtin_amdgcn_cvt_pk_bf16_f32)
  typedef __attribute__((ext_vector_type(2))) __bf16 v2bf;
  union { v2bf h; unsigned int u; } r;
  r.h = __builtin_amdgcn_cvt_pk_bf16_f32(a, b);
  return r.u;
#else
  unsigned int t0 = __float_as_uint(a) + 0x8000u;   // round-to-nearest
  unsigned int t1 = __float_as_uint(b) + 0x8000u;
  // out = { t1[31:24], t1[23:16], t0[31:24], t0[23:16] }  (s0=t1 -> bytes 4..7)
  return __builtin_amdgcn_perm(t1, t0, 0x07060302u); // v_perm_b32
#endif
}

__device__ __forceinline__ uint2 cvt4_bf16(float4 v) {
  return make_uint2(pack2_bf16(v.x, v.y), pack2_bf16(v.z, v.w));
}

union Frag {
  unsigned int u[8];
  uint4 q[2];
  v16bf v;
};

__global__ __launch_bounds__(256) void gemm_gbn_prior_kernel(
    const float* __restrict__ priors,
    const float* __restrict__ feat,
    const float* __restrict__ W,
    const float* __restrict__ gamma,
    const float* __restrict__ beta,
    float* __restrict__ out) {
  __shared__ __align__(16) unsigned int As[VBS * LDSU];  // 128 x 64 bf16 (+pad) = 18KB
  __shared__ __align__(16) unsigned int Bs[VBS * LDSU];
  __shared__ float csum[128];
  __shared__ float csum2[128];

  const int tid  = threadIdx.x;
  const int lane = tid & 31;
  const int wave = tid >> 5;
  const int wr   = wave >> 2;      // 0..1 : wave covers rows wr*64..+64
  const int wc   = wave & 3;       // 0..3 : wave covers cols wc*32..+32
  const int l16  = lane & 15;
  const int half = lane >> 4;
  const int m_base = blockIdx.x * VBS;    // ghost-batch aligned (VBS == tile M)
  const int n_base = blockIdx.y * 128;

  if (tid < 128) { csum[tid] = 0.f; csum2[tid] = 0.f; }

  const v8f vzero = {0.f, 0.f, 0.f, 0.f, 0.f, 0.f, 0.f, 0.f};
  v8f acc[4][2];
#pragma unroll
  for (int mt = 0; mt < 4; ++mt)
#pragma unroll
    for (int nt = 0; nt < 2; ++nt) acc[mt][nt] = vzero;

  // Staging: 128 rows x 64 k fp32 per operand = 2048 float4,
  // 256 threads -> 8 float4 each (16 float4 per row).
  float4 pa[8], pb[8];
#pragma unroll
  for (int i = 0; i < 8; ++i) {
    int idx = tid + i * 256;
    int r   = idx >> 4;
    int c4  = (idx & 15) * 4;
    pa[i] = *(const float4*)(feat + (size_t)(m_base + r) * IN_F + c4);
    pb[i] = *(const float4*)(W    + (size_t)(n_base + r) * IN_F + c4);
  }

  for (int kk = 0; kk < IN_F; kk += TK) {
    // convert staged registers -> packed bf16 in LDS
#pragma unroll
    for (int i = 0; i < 8; ++i) {
      int idx  = tid + i * 256;
      int r    = idx >> 4;
      int c4   = (idx & 15) * 4;
      int uoff = (r * LDSU + (c4 >> 1)) >> 1;   // uint2 index
      ((uint2*)As)[uoff] = cvt4_bf16(pa[i]);
      ((uint2*)Bs)[uoff] = cvt4_bf16(pb[i]);
    }
    __syncthreads();

    // issue next tile's global loads; they drain under the WMMAs below
    if (kk + TK < IN_F) {
#pragma unroll
      for (int i = 0; i < 8; ++i) {
        int idx = tid + i * 256;
        int r   = idx >> 4;
        int c4  = (idx & 15) * 4;
        pa[i] = *(const float4*)(feat + (size_t)(m_base + r) * IN_F + kk + TK + c4);
        pb[i] = *(const float4*)(W    + (size_t)(n_base + r) * IN_F + kk + TK + c4);
      }
    }

    // two 16x16x32 k-substeps
#pragma unroll
    for (int ks = 0; ks < 2; ++ks) {
      Frag a[4], b[2];
      // A frag: 16-bit A layout (lane: M=l16; K = half*8+{0..7}, 16+half*8+{0..7})
#pragma unroll
      for (int mt = 0; mt < 4; ++mt) {
        int r = wr * 64 + mt * 16 + l16;
        a[mt].q[0] = ((const uint4*)As)[r * 9 + ks * 4 + half];
        a[mt].q[1] = ((const uint4*)As)[r * 9 + ks * 4 + 2 + half];
      }
      // B frag: 16-bit B layout (lane: N=l16; K = half*8+{0..7}, 16+half*8+{0..7})
#pragma unroll
      for (int nt = 0; nt < 2; ++nt) {
        int r = wc * 32 + nt * 16 + l16;
        b[nt].q[0] = ((const uint4*)Bs)[r * 9 + ks * 4 + half];
        b[nt].q[1] = ((const uint4*)Bs)[r * 9 + ks * 4 + 2 + half];
      }
#pragma unroll
      for (int mt = 0; mt < 4; ++mt)
#pragma unroll
        for (int nt = 0; nt < 2; ++nt)
          acc[mt][nt] = __builtin_amdgcn_wmma_f32_16x16x32_bf16(
              false, a[mt].v, false, b[nt].v, (short)0, acc[mt][nt], false, false);
    }
    __syncthreads();
  }

  // --- fused GhostBN: per-column sum / sumsq over the 128 rows of this block ---
#pragma unroll
  for (int mt = 0; mt < 4; ++mt) {
#pragma unroll
    for (int nt = 0; nt < 2; ++nt) {
      float s = 0.f, s2 = 0.f;
#pragma unroll
      for (int i = 0; i < 8; ++i) { float v = acc[mt][nt][i]; s += v; s2 += v * v; }
      int c = wc * 32 + nt * 16 + l16;
      atomicAdd(&csum[c], s);    // ds_add_f32
      atomicAdd(&csum2[c], s2);
    }
  }
  __syncthreads();

  // --- normalize, gamma/beta, * priors, store ---
#pragma unroll
  for (int mt = 0; mt < 4; ++mt) {
#pragma unroll
    for (int nt = 0; nt < 2; ++nt) {
      int c = wc * 32 + nt * 16 + l16;
      int n = n_base + c;
      float mean = csum[c] * (1.f / VBS);
      float var  = csum2[c] * (1.f / VBS) - mean * mean;   // biased, matches ref
      float g  = gamma[n] * rsqrtf(var + 1e-5f);
      float bt = beta[n];
#pragma unroll
      for (int i = 0; i < 8; ++i) {
        int row = m_base + wr * 64 + mt * 16 + half * 8 + i;  // C-layout: VGPR i, half
        float y = (acc[mt][nt][i] - mean) * g + bt;
        y *= priors[(size_t)row * OUT_F + n];
        out[(size_t)row * OUT_F + n] = y;
      }
    }
  }
}

// ---------------------------------------------------------------------------
// Sparsemax over rows of length 1024, in place. One wave32 per row,
// 32 floats per lane in registers. Bisection on tau in [zmax-1, zmax]
// (g(t)=sum relu(z-t) is strictly decreasing), then exact finish:
// tau = (sum_{z>t} z - 1)/k  (support stable after 30 halvings).
// ---------------------------------------------------------------------------
__global__ __launch_bounds__(256) void sparsemax_kernel(float* __restrict__ out) {
  const int lane = threadIdx.x & 31;
  const int wave = threadIdx.x >> 5;
  const size_t row = (size_t)blockIdx.x * 8 + wave;
  float4* rp = (float4*)(out + row * OUT_F);

  float4 x[8];
#pragma unroll
  for (int i = 0; i < 8; ++i) x[i] = rp[i * 32 + lane];

  float zmax = -3.4e38f;
#pragma unroll
  for (int i = 0; i < 8; ++i)
    zmax = fmaxf(zmax, fmaxf(fmaxf(x[i].x, x[i].y), fmaxf(x[i].z, x[i].w)));
#pragma unroll
  for (int off = 16; off > 0; off >>= 1)
    zmax = fmaxf(zmax, __shfl_xor(zmax, off, 32));

#pragma unroll
  for (int i = 0; i < 8; ++i) {
    x[i].x -= zmax; x[i].y -= zmax; x[i].z -= zmax; x[i].w -= zmax;
  }

  float lo = -1.f, hi = 0.f;
  for (int it = 0; it < 30; ++it) {
    float t = 0.5f * (lo + hi);
    float s = 0.f;
#pragma unroll
    for (int i = 0; i < 8; ++i) {
      s += fmaxf(x[i].x - t, 0.f) + fmaxf(x[i].y - t, 0.f)
         + fmaxf(x[i].z - t, 0.f) + fmaxf(x[i].w - t, 0.f);
    }
#pragma unroll
    for (int off = 16; off > 0; off >>= 1) s += __shfl_xor(s, off, 32);
    if (s > 1.f) lo = t; else hi = t;
  }

  float t = 0.5f * (lo + hi);
  float k = 0.f, s = 0.f;
#pragma unroll
  for (int i = 0; i < 8; ++i) {
    if (x[i].x > t) { k += 1.f; s += x[i].x; }
    if (x[i].y > t) { k += 1.f; s += x[i].y; }
    if (x[i].z > t) { k += 1.f; s += x[i].z; }
    if (x[i].w > t) { k += 1.f; s += x[i].w; }
  }
#pragma unroll
  for (int off = 16; off > 0; off >>= 1) {
    k += __shfl_xor(k, off, 32);
    s += __shfl_xor(s, off, 32);
  }
  float tau = (s - 1.f) / k;

#pragma unroll
  for (int i = 0; i < 8; ++i) {
    x[i].x = fmaxf(x[i].x - tau, 0.f);
    x[i].y = fmaxf(x[i].y - tau, 0.f);
    x[i].z = fmaxf(x[i].z - tau, 0.f);
    x[i].w = fmaxf(x[i].w - tau, 0.f);
    rp[i * 32 + lane] = x[i];
  }
}

extern "C" void kernel_launch(void* const* d_in, const int* in_sizes, int n_in,
                              void* d_out, int out_size, void* d_ws, size_t ws_size,
                              hipStream_t stream) {
  const float* priors = (const float*)d_in[0];   // [16384, 1024]
  const float* feat   = (const float*)d_in[1];   // [16384, 2048]
  const float* W      = (const float*)d_in[2];   // [1024, 2048]
  const float* gamma  = (const float*)d_in[3];   // [1024]
  const float* beta   = (const float*)d_in[4];   // [1024]
  float* out = (float*)d_out;                    // [16384, 1024] f32

  dim3 grid1(NROWS / VBS, OUT_F / 128, 1);       // 128 x 8 blocks
  gemm_gbn_prior_kernel<<<grid1, 256, 0, stream>>>(priors, feat, W, gamma, beta, out);

  sparsemax_kernel<<<NROWS / 8, 256, 0, stream>>>(out);
}
